// Model_61091614818878
// MI455X (gfx1250) — matrix operand order repaired and checked by reading, not compiled, not run on hardware
//
#include <hip/hip_runtime.h>
#include <hip/hip_bf16.h>
#include <math.h>

// ---------------------------------------------------------------------------
// Longformer-ish encoder forward for MI455X (gfx1250, wave32, WMMA).
// - all dense GEMMs + band attention via v_wmma_f32_16x16x32_bf16
// - K tiles staged to LDS with global_load_async_to_lds_b128 (ASYNCcnt),
//   double-buffered so the fetch of tile k+1 overlaps WMMA on tile k
// ---------------------------------------------------------------------------

typedef __attribute__((ext_vector_type(16))) __bf16 v16bf;
typedef __attribute__((ext_vector_type(8)))  float  v8f;

union FragBF {            // one WMMA A/B operand: 8 VGPRs = 16 bf16 per lane
    unsigned int u[8];
    __bf16       e[16];
    v16bf        v;
};

constexpr int CB  = 2;
constexpr int CS  = 4096;
constexpr int CD  = 768;
constexpr int CH  = 12;
constexpr int CDH = 64;
constexpr int CW  = 256;
constexpr int CFF = 3072;
constexpr int CL  = 4;
constexpr int CNC = CS / CW;     // 16
constexpr int CKW = 3 * CW;      // 768 band width
constexpr int CBS = CB * CS;     // 8192 rows
constexpr float NEGBIG = -1000000000.0f;
constexpr float SCALE  = 0.125f; // 1/sqrt(64)

// --- gfx1250 async global->LDS copy (ASYNCcnt), 16B per lane -----------------
// ISA 10.2: for LDS-aperture flat addresses, LDS byte address = addr[31:0].
__device__ __forceinline__ void cp_async16(void* lds, const void* g)
{
#if defined(__gfx1250__)
    unsigned loff = (unsigned)(unsigned long long)lds;
    asm volatile("global_load_async_to_lds_b128 %0, %1, off"
                 :: "v"(loff), "v"(g) : "memory");
#else
    *(uint4*)lds = *(const uint4*)g;
#endif
}
__device__ __forceinline__ void cp_async_wait()
{
#if defined(__gfx1250__)
    asm volatile("s_wait_asynccnt 0x0" ::: "memory");
#endif
}

// ---------------------------------------------------------------------------
// Generic bf16 WMMA GEMM: C[M,N] = act(A[M,K] @ B[K,N] + bias)
// 256 threads = 8 waves; block tile 128x128; wave tile 32x64; K step 32.
// Double-buffered LDS, async staged. M%128==0, N%128==0, K%32==0.
// act: 0 = none, 1 = exact GELU, 2 = tanh
// ---------------------------------------------------------------------------
__global__ __launch_bounds__(256)
void gemm_bf16_wmma(const __bf16* __restrict__ A, const __bf16* __restrict__ Bm,
                    const float* __restrict__ bias, float* __restrict__ Cf,
                    __bf16* __restrict__ Cb, int M, int N, int K, int act)
{
    __shared__ unsigned int AsU[2][2048];   // 2 x (128 rows x 32 bf16) = 16 KB
    __shared__ unsigned int BsU[2][2048];   // 2 x (32 rows x 128 bf16) = 16 KB

    const unsigned int* Ag = (const unsigned int*)A;
    const unsigned int* Bg = (const unsigned int*)Bm;

    const int bm = blockIdx.y * 128, bn = blockIdx.x * 128;
    const int tid = threadIdx.x, lane = tid & 31, wave = tid >> 5;
    const int wr = wave >> 1, wc = wave & 1;   // 4x2 wave grid
    const int Ku = K >> 1, Nu = N >> 1;

    // stage one 128x32 A tile + 32x128 B tile (each thread: 2x16B per tile)
    auto stage = [&](int buf, int k0) {
#pragma unroll
        for (int it = 0; it < 2; ++it) {
            int ch = tid + it * 256;                 // 16B chunk id, 0..511
            int ra = ch >> 2, ca = (ch & 3) * 4;
            cp_async16(&AsU[buf][ch * 4],
                       Ag + (size_t)(bm + ra) * Ku + (k0 >> 1) + ca);
            int rb = ch >> 4, cb = (ch & 15) * 4;
            cp_async16(&BsU[buf][ch * 4],
                       Bg + (size_t)(k0 + rb) * Nu + (bn >> 1) + cb);
        }
    };

    const v8f vzero = {0.f, 0.f, 0.f, 0.f, 0.f, 0.f, 0.f, 0.f};
    v8f acc[2][4];
#pragma unroll
    for (int i = 0; i < 2; ++i)
#pragma unroll
        for (int j = 0; j < 4; ++j) acc[i][j] = vzero;

    stage(0, 0);
    cp_async_wait();
    __syncthreads();

    int buf = 0;
    for (int k0 = 0; k0 < K; k0 += 32) {
        if (k0 + 32 < K) stage(buf ^ 1, k0 + 32);      // overlap with WMMA below
        if (k0 + 64 < K) {                             // warm WGP$ two tiles out
            __builtin_prefetch(Ag + (size_t)(bm + (tid >> 2)) * Ku + ((k0 + 64) >> 1) + (tid & 3) * 4, 0, 3);
            __builtin_prefetch(Bg + (size_t)(k0 + 64 + (tid >> 4)) * Nu + (bn >> 1) + (tid & 15) * 4, 0, 3);
        }

        const unsigned int* As = AsU[buf];
        const unsigned int* Bs = BsU[buf];

        // A fragments (ISA 16-bit 16x32 layout: lanes<16 hold K 0..7/16..23)
        const int aoff = (lane & 16) ? 4 : 0;
        const int arow = wr * 32 + (lane & 15);
        FragBF afr[2];
#pragma unroll
        for (int i = 0; i < 2; ++i) {
            const unsigned int* p = As + (arow + i * 16) * 16;
            *(uint4*)&afr[i].u[0] = *(const uint4*)(p + aoff);
            *(uint4*)&afr[i].u[4] = *(const uint4*)(p + 8 + aoff);
        }
#pragma unroll
        for (int jt = 0; jt < 4; ++jt) {
            FragBF bfr;                     // lane%16 = col, elements over K (table order)
            const __bf16* pb = (const __bf16*)Bs + wc * 64 + jt * 16 + (lane & 15);
#pragma unroll
            for (int s = 0; s < 16; ++s)
                bfr.e[s] = pb[(s < 8 ? ((lane >> 4) * 8 + s) : (16 + (lane >> 4) * 8 + s - 8)) * 128];
#pragma unroll
            for (int i = 0; i < 2; ++i)
                acc[i][jt] = __builtin_amdgcn_wmma_f32_16x16x32_bf16(
                    false, afr[i].v, false, bfr.v, (short)0, acc[i][jt], false, false);
        }
        cp_async_wait();
        __syncthreads();
        buf ^= 1;
    }

    // C layout: lanes<16 -> N=lane, M rows 0..7 in v0..7; lanes>=16 -> M rows 8..15
    const int cl = lane & 15;
    const int rbase = (lane & 16) ? 8 : 0;
#pragma unroll
    for (int i = 0; i < 2; ++i) {
#pragma unroll
        for (int jt = 0; jt < 4; ++jt) {
            int col = bn + wc * 64 + jt * 16 + cl;
            float bv = bias ? bias[col] : 0.f;
#pragma unroll
            for (int r = 0; r < 8; ++r) {
                int row = bm + wr * 32 + i * 16 + rbase + r;
                float vv = acc[i][jt][r] + bv;
                if (act == 1)      vv = 0.5f * vv * (1.f + erff(vv * 0.70710678f));
                else if (act == 2) vv = tanhf(vv);
                size_t o = (size_t)row * N + col;
                if (Cf) Cf[o] = vv;
                if (Cb) Cb[o] = (__bf16)vv;
            }
        }
    }
}

// ---------------------------------------------------------------------------
// Band attention scores: per (chunk,h,b) block, scores[256 x 768] = q @ kc^T
// k chunk staged TRANSPOSED into LDS (kcT[d][j], padded rows) in two halves,
// each WMMA B fragment gathers 16 bf16 at row stride KTP2. q,k bf16 as [b, s, h, dh].
// ---------------------------------------------------------------------------
constexpr int KTP2 = 392;    // padded LDS row stride (bf16) for 384 columns

__global__ __launch_bounds__(256)
void attn_scores(const __bf16* __restrict__ qb, const __bf16* __restrict__ kb,
                 const int* __restrict__ pad, float* __restrict__ scores)
{
    __shared__ __bf16 kcT[64 * KTP2];                  // ~49 KB

    const int c = blockIdx.x, h = blockIdx.y, b = blockIdx.z;
    const int lane = threadIdx.x & 31, wave = threadIdx.x >> 5;
    const unsigned int* qU = (const unsigned int*)qb;
    const int kbase = c * CW - CW;

    // A fragments: 2 row tiles x 2 K steps, direct 128-bit global loads
    const int aoff = (lane & 16) ? 4 : 0;
    FragBF afr[2][2];
#pragma unroll
    for (int i = 0; i < 2; ++i) {
        int s = c * CW + wave * 32 + i * 16 + (lane & 15);
        size_t rb = ((size_t)(b * CS + s) * CH + h) * 32;    // u32 per q row
#pragma unroll
        for (int kk = 0; kk < 2; ++kk) {
            const unsigned int* p = qU + rb + kk * 16;
            *(uint4*)&afr[i][kk].u[0] = *(const uint4*)(p + aoff);
            *(uint4*)&afr[i][kk].u[4] = *(const uint4*)(p + 8 + aoff);
        }
    }

    const v8f vzero = {0.f, 0.f, 0.f, 0.f, 0.f, 0.f, 0.f, 0.f};
    const int cl = lane & 15, rbase = (lane & 16) ? 8 : 0;

    for (int half = 0; half < 2; ++half) {
        __syncthreads();                    // previous half's reads complete
        // fill kcT[d][jl] for j in [half*384, half*384+384)
        for (int it = 0; it < 12; ++it) {
            int ch = threadIdx.x + it * 256;          // 0..3071: jl(384) x dg(8)
            int jl = ch >> 3, dg = ch & 7;
            int kpos = kbase + half * 384 + jl;
            uint4 val = make_uint4(0u, 0u, 0u, 0u);
            if (kpos >= 0 && kpos < CS)
                val = *(const uint4*)(kb + ((size_t)(b * CS + kpos) * CH + h) * CDH + dg * 8);
            const __bf16* e = (const __bf16*)&val;
#pragma unroll
            for (int t = 0; t < 8; ++t) kcT[(dg * 8 + t) * KTP2 + jl] = e[t];
        }
        __syncthreads();

        for (int jt = half * 24; jt < half * 24 + 24; ++jt) {
            v8f acc0 = vzero, acc1 = vzero;
#pragma unroll
            for (int kk = 0; kk < 2; ++kk) {
                FragBF bfr;                 // B[k=d][n=j] = kcT[d][j]
                const __bf16* src = &kcT[(kk * 32) * KTP2 + (jt * 16 - half * 384) + (lane & 15)];
#pragma unroll
                for (int s = 0; s < 16; ++s)
                    bfr.e[s] = src[(s < 8 ? ((lane >> 4) * 8 + s) : (16 + (lane >> 4) * 8 + s - 8)) * KTP2];
                acc0 = __builtin_amdgcn_wmma_f32_16x16x32_bf16(
                    false, afr[0][kk].v, false, bfr.v, (short)0, acc0, false, false);
                acc1 = __builtin_amdgcn_wmma_f32_16x16x32_bf16(
                    false, afr[1][kk].v, false, bfr.v, (short)0, acc1, false, false);
            }
            const int j = jt * 16 + cl;
            const int kpos = kbase + j;
            const int kcl = kpos < 0 ? 0 : (kpos >= CS ? CS - 1 : kpos);
            const bool kok = (kpos >= 0) && (kpos < CS) && (pad[b * CS + kcl] > 0);
#pragma unroll
            for (int i = 0; i < 2; ++i) {
                v8f a = i ? acc1 : acc0;
#pragma unroll
                for (int r = 0; r < 8; ++r) {
                    int irow = wave * 32 + i * 16 + rbase + r;
                    int rel = j - CW - irow;
                    float vv = (kok && rel >= -CW && rel <= CW) ? a[r] * SCALE : NEGBIG;
                    scores[((size_t)((b * CH + h) * CS) + c * CW + irow) * CKW + j] = vv;
                }
            }
        }
    }
}

// global-attention column score: gcol[b,h,s] = scale * q[b,s,h]·kg[b,0,h]
__global__ void gcol_kernel(const __bf16* __restrict__ qb,
                            const __bf16* __restrict__ kgb, float* __restrict__ gcol)
{
    int idx = blockIdx.x * 256 + threadIdx.x;
    if (idx >= CB * CH * CS) return;
    int s = idx % CS, bh = idx / CS, h = bh % CH, b = bh / CH;
    const __bf16* q  = qb  + ((size_t)(b * CS + s) * CH + h) * CDH;
    const __bf16* kg = kgb + ((size_t)(b * CS + 0) * CH + h) * CDH;
    float dot = 0.f;
    for (int d = 0; d < CDH; ++d) dot += (float)q[d] * (float)kg[d];
    gcol[idx] = dot * SCALE;
}

// softmax over [gcol, band(768)]; one wave per row
__global__ __launch_bounds__(256)
void band_softmax(const float* __restrict__ scores, const float* __restrict__ gcol,
                  __bf16* __restrict__ probs, float* __restrict__ p0)
{
    int row  = blockIdx.x * 8 + (threadIdx.x >> 5);
    int lane = threadIdx.x & 31;
    const float* sr = scores + (size_t)row * CKW;
    float g = gcol[row];
    float vals[24];
    float mx = g;
#pragma unroll
    for (int t = 0; t < 24; ++t) { vals[t] = sr[lane + 32 * t]; mx = fmaxf(mx, vals[t]); }
#pragma unroll
    for (int o = 16; o; o >>= 1) mx = fmaxf(mx, __shfl_xor(mx, o, 32));
    float ge  = __expf(g - mx);
    float sum = (lane == 0) ? ge : 0.f;
#pragma unroll
    for (int t = 0; t < 24; ++t) { vals[t] = __expf(vals[t] - mx); sum += vals[t]; }
#pragma unroll
    for (int o = 16; o; o >>= 1) sum += __shfl_xor(sum, o, 32);
    float inv = 1.f / sum;
    __bf16* pr = probs + (size_t)row * CKW;
#pragma unroll
    for (int t = 0; t < 24; ++t) pr[lane + 32 * t] = (__bf16)(vals[t] * inv);
    if (lane == 0) p0[row] = ge * inv;
}

// ---------------------------------------------------------------------------
// ctx = probs[256x768] @ vc[768x64] + p0 * vg0 ; per (chunk,h,b) block; WMMA
// output written transposed into [b, s, h, dh] bf16 (ready for the Wo GEMM)
// ---------------------------------------------------------------------------
__global__ __launch_bounds__(256)
void attn_ctx(const __bf16* __restrict__ probs, const __bf16* __restrict__ vb,
              const __bf16* __restrict__ vgb, const float* __restrict__ p0,
              __bf16* __restrict__ ctxb)
{
    const int c = blockIdx.x, h = blockIdx.y, b = blockIdx.z;
    const int lane = threadIdx.x & 31, wave = threadIdx.x >> 5;
    const unsigned int* pU = (const unsigned int*)probs;
    const unsigned int* vU = (const unsigned int*)vb;

    const v8f vzero = {0.f, 0.f, 0.f, 0.f, 0.f, 0.f, 0.f, 0.f};
    v8f acc[2][4];
#pragma unroll
    for (int i = 0; i < 2; ++i)
#pragma unroll
        for (int j = 0; j < 4; ++j) acc[i][j] = vzero;

    const int aoff = (lane & 16) ? 4 : 0;
    size_t rowbase[2];
#pragma unroll
    for (int i = 0; i < 2; ++i) {
        int irow = wave * 32 + i * 16 + (lane & 15);
        rowbase[i] = ((size_t)(b * CH + h) * CS + c * CW + irow) * 384;  // u32/row
    }

    for (int k0 = 0; k0 < CKW; k0 += 32) {
        FragBF afr[2];
#pragma unroll
        for (int i = 0; i < 2; ++i) {
            const unsigned int* p = pU + rowbase[i] + (k0 >> 1);
            *(uint4*)&afr[i].u[0] = *(const uint4*)(p + aoff);
            *(uint4*)&afr[i].u[4] = *(const uint4*)(p + 8 + aoff);
        }
        int kpos = c * CW - CW + k0 + lane;
        bool ok  = (kpos >= 0) && (kpos < CS);
        size_t vrow = ((size_t)(b * CS + (ok ? kpos : 0)) * CH + h) * 32;  // u32
#pragma unroll
        for (int jt = 0; jt < 4; ++jt) {
            FragBF bfr;
#pragma unroll
            for (int s = 0; s < 16; ++s) {
                int kp = c * CW - CW + k0 + (s < 8 ? ((lane >> 4) * 8 + s) : (16 + (lane >> 4) * 8 + s - 8));
                bfr.e[s] = (kp >= 0 && kp < CS)
                    ? vb[((size_t)(b * CS + kp) * CH + h) * CDH + jt * 16 + (lane & 15)]
                    : (__bf16)0.f;
            }
#pragma unroll
            for (int i = 0; i < 2; ++i)
                acc[i][jt] = __builtin_amdgcn_wmma_f32_16x16x32_bf16(
                    false, afr[i].v, false, bfr.v, (short)0, acc[i][jt], false, false);
        }
    }

    const int cl = lane & 15, rbase = (lane & 16) ? 8 : 0;
#pragma unroll
    for (int i = 0; i < 2; ++i) {
#pragma unroll
        for (int jt = 0; jt < 4; ++jt) {
            int d = jt * 16 + cl;
            float vg0 = (float)vgb[((size_t)(b * CS) * CH + h) * CDH + d];
#pragma unroll
            for (int r = 0; r < 8; ++r) {
                int irow = wave * 32 + i * 16 + rbase + r;
                int s = c * CW + irow;
                float pz = p0[(size_t)(b * CH + h) * CS + s];
                float vv = acc[i][jt][r] + pz * vg0;
                ctxb[((size_t)(b * CS + s) * CH + h) * CDH + d] = (__bf16)vv;
            }
        }
    }
}

// global attention for token 0: ctx0 = softmax(qg0·kg / sqrt(dh)) @ vg
__global__ __launch_bounds__(256)
void attn_global(const __bf16* __restrict__ qgb, const __bf16* __restrict__ kgb,
                 const __bf16* __restrict__ vgb, const int* __restrict__ pad,
                 __bf16* __restrict__ ctxb)
{
    const int h = blockIdx.x, b = blockIdx.y;
    __shared__ float qg0[64];
    __shared__ float red[8];
    __shared__ float ctx[64];
    if (threadIdx.x < 64) {
        qg0[threadIdx.x] = (float)qgb[((size_t)(b * CS) * CH + h) * CDH + threadIdx.x];
        ctx[threadIdx.x] = 0.f;
    }
    __syncthreads();

    float sc[16];
    float mx = -3e38f;
    for (int t = 0; t < 16; ++t) {
        int s = threadIdx.x + t * 256;
        const __bf16* kr = kgb + ((size_t)(b * CS + s) * CH + h) * CDH;
        float dot = 0.f;
        for (int d = 0; d < CDH; ++d) dot += qg0[d] * (float)kr[d];
        dot = (pad[b * CS + s] > 0) ? dot * SCALE : NEGBIG;
        sc[t] = dot;
        mx = fmaxf(mx, dot);
    }
    int lane = threadIdx.x & 31, w = threadIdx.x >> 5;
    for (int o = 16; o; o >>= 1) mx = fmaxf(mx, __shfl_xor(mx, o, 32));
    if (lane == 0) red[w] = mx;
    __syncthreads();
    mx = red[0];
    for (int i = 1; i < 8; ++i) mx = fmaxf(mx, red[i]);
    float sum = 0.f;
    for (int t = 0; t < 16; ++t) { sc[t] = __expf(sc[t] - mx); sum += sc[t]; }
    for (int o = 16; o; o >>= 1) sum += __shfl_xor(sum, o, 32);
    __syncthreads();
    if (lane == 0) red[w] = sum;
    __syncthreads();
    sum = 0.f;
    for (int i = 0; i < 8; ++i) sum += red[i];
    float inv = 1.f / sum;

    for (int dc = 0; dc < 4; ++dc) {
        float accv[16];
        for (int dd = 0; dd < 16; ++dd) accv[dd] = 0.f;
        for (int t = 0; t < 16; ++t) {
            int s = threadIdx.x + t * 256;
            float p = sc[t] * inv;
            const __bf16* vr = vgb + ((size_t)(b * CS + s) * CH + h) * CDH + dc * 16;
            for (int dd = 0; dd < 16; ++dd) accv[dd] += p * (float)vr[dd];
        }
        for (int dd = 0; dd < 16; ++dd) atomicAdd(&ctx[dc * 16 + dd], accv[dd]);
    }
    __syncthreads();
    if (threadIdx.x < 64)
        ctxb[((size_t)(b * CS) * CH + h) * CDH + threadIdx.x] = (__bf16)ctx[threadIdx.x];
}

// ---------------------------------------------------------------------------
// LayerNorm over D=768 of (x [+ y]); one block per row; writes f32 + bf16
// ---------------------------------------------------------------------------
__global__ __launch_bounds__(256)
void row_ln(const float* __restrict__ x, const float* __restrict__ y,
            const float* __restrict__ g, const float* __restrict__ be,
            float* __restrict__ of, __bf16* __restrict__ ob)
{
    int row = blockIdx.x;
    const float* xr = x + (size_t)row * CD;
    const float* yr = y ? (y + (size_t)row * CD) : nullptr;
    float v3[3];
    float sum = 0.f;
#pragma unroll
    for (int i = 0; i < 3; ++i) {
        int d = threadIdx.x + i * 256;
        float vv = xr[d] + (yr ? yr[d] : 0.f);
        v3[i] = vv;
        sum += vv;
    }
    __shared__ float red[8];
    __shared__ float stat[2];
    int lane = threadIdx.x & 31, w = threadIdx.x >> 5;
    for (int o = 16; o; o >>= 1) sum += __shfl_xor(sum, o, 32);
    if (lane == 0) red[w] = sum;
    __syncthreads();
    if (threadIdx.x == 0) {
        float s = 0.f;
        for (int i = 0; i < 8; ++i) s += red[i];
        stat[0] = s * (1.f / CD);
    }
    __syncthreads();
    float mean = stat[0];
    float vs = 0.f;
#pragma unroll
    for (int i = 0; i < 3; ++i) { float d0 = v3[i] - mean; vs += d0 * d0; }
    for (int o = 16; o; o >>= 1) vs += __shfl_xor(vs, o, 32);
    __syncthreads();
    if (lane == 0) red[w] = vs;
    __syncthreads();
    if (threadIdx.x == 0) {
        float s = 0.f;
        for (int i = 0; i < 8; ++i) s += red[i];
        stat[1] = rsqrtf(s * (1.f / CD) + 1e-5f);
    }
    __syncthreads();
    float inv = stat[1];
#pragma unroll
    for (int i = 0; i < 3; ++i) {
        int d = threadIdx.x + i * 256;
        float o = g[d] * (v3[i] - mean) * inv + be[d];
        of[(size_t)row * CD + d] = o;
        if (ob) ob[(size_t)row * CD + d] = (__bf16)o;
    }
}

__global__ void embed_gather(const int* __restrict__ x, const float* __restrict__ ew,
                             const float* __restrict__ ep, const float* __restrict__ et,
                             float* __restrict__ hf)
{
    size_t idx = (size_t)blockIdx.x * 256 + threadIdx.x;
    if (idx >= (size_t)CBS * CD) return;
    int d = (int)(idx % CD);
    int row = (int)(idx / CD);
    int s = row % CS;
    int tok = x[row];
    hf[idx] = ew[(size_t)tok * CD + d] + ep[(size_t)(s + 2) * CD + d] + et[d];
}

__global__ void to_bf16(const float* __restrict__ in, __bf16* __restrict__ out, size_t n)
{
    size_t i = (size_t)blockIdx.x * 256 + threadIdx.x;
    size_t stride = (size_t)gridDim.x * 256;
    for (; i < n; i += stride) out[i] = (__bf16)in[i];
}

// classifier: t = tanh(h[:,0] @ Wd + bd)
__global__ void cls_dense(const float* __restrict__ hf, const float* __restrict__ Wd,
                          const float* __restrict__ bd, float* __restrict__ t)
{
    int idx = blockIdx.x * 256 + threadIdx.x;
    if (idx >= CB * CD) return;
    int bb = idx / CD, j = idx % CD;
    const float* h0 = hf + (size_t)bb * CS * CD;
    float s = bd[j];
    for (int d = 0; d < CD; ++d) s += h0[d] * Wd[(size_t)d * CD + j];
    t[idx] = tanhf(s);
}

// logits + log-softmax + NLL loss; d_out = [loss, logits(2x2)]
__global__ void cls_out(const float* __restrict__ t, const float* __restrict__ Wo,
                        const float* __restrict__ bo, const int* __restrict__ target,
                        float* __restrict__ out)
{
    __shared__ float lg[4];
    int idx = threadIdx.x;
    if (idx < 4) {
        int bb = idx >> 1, c = idx & 1;
        float s = bo[c];
        for (int d = 0; d < CD; ++d) s += t[bb * CD + d] * Wo[d * 2 + c];
        lg[idx] = s;
    }
    __syncthreads();
    if (idx == 0) {
        float loss = 0.f;
        for (int bb = 0; bb < 2; ++bb) {
            float m = fmaxf(lg[bb * 2], lg[bb * 2 + 1]);
            float z = __expf(lg[bb * 2] - m) + __expf(lg[bb * 2 + 1] - m);
            float lse = m + logf(z);
            loss += -(lg[bb * 2 + target[bb]] - lse);
            out[1 + bb * 2 + 0] = lg[bb * 2 + 0];
            out[1 + bb * 2 + 1] = lg[bb * 2 + 1];
        }
        out[0] = loss * 0.5f;
    }
}

// ---------------------------------------------------------------------------
// host orchestration
// ---------------------------------------------------------------------------
extern "C" void kernel_launch(void* const* d_in, const int* in_sizes, int n_in,
                              void* d_out, int out_size, void* d_ws, size_t ws_size,
                              hipStream_t stream)
{
    (void)in_sizes; (void)n_in; (void)out_size; (void)ws_size;

    const int*   x       = (const int*)d_in[0];
    const int*   target  = (const int*)d_in[1];
    const int*   pad     = (const int*)d_in[2];
    const float* emb_w   = (const float*)d_in[3];
    const float* emb_p   = (const float*)d_in[4];
    const float* emb_t   = (const float*)d_in[5];
    const float* eln_g   = (const float*)d_in[6];
    const float* eln_b   = (const float*)d_in[7];
    const float* cls_Wd  = (const float*)d_in[8];
    const float* cls_bd  = (const float*)d_in[9];
    const float* cls_Wo  = (const float*)d_in[10];
    const float* cls_bo  = (const float*)d_in[11];
    const int wIdx[7] = {12, 14, 16, 18, 20, 22, 24};   // Wq..Wvg, Wo (each L*D*D)
    const int bIdx[7] = {13, 15, 17, 19, 21, 23, 25};
    const float* Wi_f = (const float*)d_in[26];
    const float* bi_f = (const float*)d_in[27];
    const float* Wf_f = (const float*)d_in[28];
    const float* bf_f = (const float*)d_in[29];
    const float* ln1g = (const float*)d_in[30];
    const float* ln1b = (const float*)d_in[31];
    const float* ln2g = (const float*)d_in[32];
    const float* ln2b = (const float*)d_in[33];

    // ---- workspace carve (256B aligned bump allocator) ----
    char* wsp = (char*)d_ws;
    size_t off = 0;
    auto alloc = [&](size_t bytes) -> void* {
        void* p = wsp + off;
        off = (off + bytes + 255) & ~(size_t)255;
        return p;
    };
    const size_t NT = (size_t)CBS * CD;            // tokens*dim
    float*  hf    = (float*)alloc(NT * 4);
    __bf16* hb    = (__bf16*)alloc(NT * 2);
    float*  h1f   = (float*)alloc(NT * 4);
    __bf16* h1b   = (__bf16*)alloc(NT * 2);
    __bf16* qb2   = (__bf16*)alloc(NT * 2);
    __bf16* kb2   = (__bf16*)alloc(NT * 2);
    __bf16* vb2   = (__bf16*)alloc(NT * 2);
    __bf16* qgb   = (__bf16*)alloc(NT * 2);
    __bf16* kgb   = (__bf16*)alloc(NT * 2);
    __bf16* vgb   = (__bf16*)alloc(NT * 2);
    __bf16* ctxb  = (__bf16*)alloc(NT * 2);
    float*  af    = (float*)alloc(NT * 4);
    float*  ffo   = (float*)alloc(NT * 4);
    __bf16* fb2   = (__bf16*)alloc((size_t)CBS * CFF * 2);
    float*  scores= (float*)alloc((size_t)CB * CH * CS * CKW * 4);
    __bf16* probs = (__bf16*)alloc((size_t)CB * CH * CS * CKW * 2);
    float*  gcolB = (float*)alloc((size_t)CB * CH * CS * 4);
    float*  p0B   = (float*)alloc((size_t)CB * CH * CS * 4);
    float*  tbuf  = (float*)alloc((size_t)CB * CD * 4);
    __bf16* WprojB[7];
    for (int i = 0; i < 7; ++i) WprojB[i] = (__bf16*)alloc((size_t)CL * CD * CD * 2);
    __bf16* WiB = (__bf16*)alloc((size_t)CL * CD * CFF * 2);
    __bf16* WfB = (__bf16*)alloc((size_t)CL * CFF * CD * 2);

    // ---- weight conversion to bf16 (weights stay resident in 192MB L2) ----
    for (int i = 0; i < 7; ++i)
        to_bf16<<<4096, 256, 0, stream>>>((const float*)d_in[wIdx[i]], WprojB[i],
                                          (size_t)CL * CD * CD);
    to_bf16<<<8192, 256, 0, stream>>>(Wi_f, WiB, (size_t)CL * CD * CFF);
    to_bf16<<<8192, 256, 0, stream>>>(Wf_f, WfB, (size_t)CL * CFF * CD);

    // ---- embedding + LN ----
    embed_gather<<<(int)((NT + 255) / 256), 256, 0, stream>>>(x, emb_w, emb_p, emb_t, hf);
    row_ln<<<CBS, 256, 0, stream>>>(hf, nullptr, eln_g, eln_b, hf, hb);

    auto gemm = [&](const __bf16* A, const __bf16* Bm, const float* bias,
                    float* Cf, __bf16* Cb, int M, int N, int K, int act) {
        dim3 grid(N / 128, M / 128);
        gemm_bf16_wmma<<<grid, 256, 0, stream>>>(A, Bm, bias, Cf, Cb, M, N, K, act);
    };

    // ---- transformer layers ----
    for (int l = 0; l < CL; ++l) {
        const size_t wo = (size_t)l * CD * CD;
        __bf16* proj[6] = {qb2, kb2, vb2, qgb, kgb, vgb};
        for (int p = 0; p < 6; ++p)
            gemm(hb, WprojB[p] + wo, (const float*)d_in[bIdx[p]] + (size_t)l * CD,
                 nullptr, proj[p], CBS, CD, CD, 0);

        gcol_kernel<<<(CB * CH * CS + 255) / 256, 256, 0, stream>>>(qb2, kgb, gcolB);
        attn_scores<<<dim3(CNC, CH, CB), 256, 0, stream>>>(qb2, kb2, pad, scores);
        band_softmax<<<CB * CH * CS / 8, 256, 0, stream>>>(scores, gcolB, probs, p0B);
        attn_ctx<<<dim3(CNC, CH, CB), 256, 0, stream>>>(probs, vb2, vgb, p0B, ctxb);
        attn_global<<<dim3(CH, CB), 256, 0, stream>>>(qgb, kgb, vgb, pad, ctxb);

        gemm(ctxb, WprojB[6] + wo, (const float*)d_in[bIdx[6]] + (size_t)l * CD,
             af, nullptr, CBS, CD, CD, 0);
        row_ln<<<CBS, 256, 0, stream>>>(hf, af, ln1g + (size_t)l * CD,
                                        ln1b + (size_t)l * CD, h1f, h1b);

        gemm(h1b, WiB + (size_t)l * CD * CFF, bi_f + (size_t)l * CFF,
             nullptr, fb2, CBS, CFF, CD, 1 /*gelu*/);
        gemm(fb2, WfB + (size_t)l * CFF * CD, bf_f + (size_t)l * CD,
             ffo, nullptr, CBS, CD, CFF, 0);
        row_ln<<<CBS, 256, 0, stream>>>(h1f, ffo, ln2g + (size_t)l * CD,
                                        ln2b + (size_t)l * CD, hf, hb);
    }

    // ---- classifier + loss ----
    cls_dense<<<(CB * CD + 255) / 256, 256, 0, stream>>>(hf, cls_Wd, cls_bd, tbuf);
    cls_out<<<1, 64, 0, stream>>>(tbuf, cls_Wo, cls_bo, target, (float*)d_out);
}
